// AttentionBlock_13743895347718
// MI455X (gfx1250) — matrix operand order repaired and checked by reading, not compiled, not run on hardware
//
#include <hip/hip_runtime.h>

// Problem constants (from reference)
#define D_B   8
#define NT_   2048
#define NDIM  64
#define N_H   8
#define CHUNK 64
#define NC    (NT_ / CHUNK)   // 32 chunks
#define LDK   72              // LDS row stride in bf16 elems (144B, 16B-aligned rows)

typedef __attribute__((ext_vector_type(16))) __bf16 v16bf;
typedef __attribute__((ext_vector_type(8)))  __bf16 bf16x8;
typedef __attribute__((ext_vector_type(8)))  float  v8f;

// ---------------- WMMA fragment helpers (layouts per CDNA5 ISA 7.12.2) ----------------

// A fragment: 16x32 bf16 tile from row-major LDS matrix (row stride LDK).
// lane: m = lane&15 ; hi = lane>>4 ; elems 0..7 -> k = k0+8*hi+e ; elems 8..15 -> k0+16+8*hi+e
__device__ __forceinline__ v16bf frag_a(const __bf16* base, int m0, int k0, int lane) {
  const int m = lane & 15, hi = lane >> 4;
  const __bf16* p = base + (m0 + m) * LDK + k0 + 8 * hi;
  bf16x8 lo = *(const bf16x8*)(p);
  bf16x8 hh = *(const bf16x8*)(p + 16);
  v16bf r;
#pragma unroll
  for (int e = 0; e < 8; ++e) { r[e] = lo[e]; r[e + 8] = hh[e]; }
  return r;
}

// B fragment: 32x16 bf16 tile, sourced from an N-major image Bt[n][k] (row stride LDK).
// lane: n = lane&15 ; hi = lane>>4 ; elems 0..7 -> k = k0+8*hi+e ; elems 8..15 -> k0+16+8*hi+e
__device__ __forceinline__ v16bf frag_bt(const __bf16* base, int n0, int k0, int lane) {
  const int n = lane & 15, hi = lane >> 4;
  const __bf16* p = base + (n0 + n) * LDK + k0 + 8 * hi;
  bf16x8 lo = *(const bf16x8*)(p);
  bf16x8 hh = *(const bf16x8*)(p + 16);
  v16bf r;
#pragma unroll
  for (int e = 0; e < 8; ++e) { r[e] = lo[e]; r[e + 8] = hh[e]; }
  return r;
}

__device__ __forceinline__ v8f wmma_bf16(v16bf a, v16bf b, v8f c) {
  return __builtin_amdgcn_wmma_f32_16x16x32_bf16(false, a, false, b, (short)0, c, false, false);
}

// D layout: element r of lane -> (M = r + 8*hi, N = lane&15)
__device__ __forceinline__ void store_bf16(__bf16* M, int mt, int nt, int lane, v8f d) {
  const int n = lane & 15, hi = lane >> 4;
#pragma unroll
  for (int r = 0; r < 8; ++r)
    M[(mt * 16 + r + 8 * hi) * LDK + nt * 16 + n] = (__bf16)d[r];
}
__device__ __forceinline__ void store_bf16_t(__bf16* M, int mt, int nt, int lane, v8f d) {
  const int n = lane & 15, hi = lane >> 4;
#pragma unroll
  for (int r = 0; r < 8; ++r)
    M[(nt * 16 + n) * LDK + mt * 16 + r + 8 * hi] = (__bf16)d[r];
}

// ---------------- Kernel 1: per-chunk Gram matrices G_c = R_c^T R_c ----------------
__global__ void __launch_bounds__(256)
g_kernel(const float* __restrict__ rp, float* __restrict__ G) {
  alignas(16) __shared__ __bf16 Rt[64 * LDK];   // R^T : [i][t]
  const int bc = blockIdx.x, b = bc / NC, c = bc % NC;
  const int tid = threadIdx.x;
  const float* Rg = rp + ((size_t)b * NT_ + (size_t)c * CHUNK) * NDIM;

  for (int idx = tid; idx < NDIM * NDIM; idx += 256) {
    const int t = idx >> 6, i = idx & 63;
    Rt[i * LDK + t] = (__bf16)Rg[idx];
  }
  __syncthreads();

  const int wave = tid >> 5, lane = tid & 31;
  const int mt = wave & 3, ntb = (wave >> 2) * 2;
  float* Gp = G + ((size_t)b * NC + c) * (NDIM * NDIM);
  const v8f vz = {0.f, 0.f, 0.f, 0.f, 0.f, 0.f, 0.f, 0.f};

  // G[i,j] = sum_t R[t,i]R[t,j] : A = Rt (i x t), Bt image = Rt (j x t)
  v8f acc[2] = {vz, vz};
#pragma unroll
  for (int k0 = 0; k0 < 64; k0 += 32) {
    v16bf a = frag_a(Rt, mt * 16, k0, lane);          // shared across both n-tiles
#pragma unroll
    for (int s = 0; s < 2; ++s)
      acc[s] = wmma_bf16(a, frag_bt(Rt, (ntb + s) * 16, k0, lane), acc[s]);
  }
  const int n = lane & 15, hi = lane >> 4;
#pragma unroll
  for (int s = 0; s < 2; ++s)
#pragma unroll
    for (int r = 0; r < 8; ++r)
      Gp[(mt * 16 + r + 8 * hi) * NDIM + (ntb + s) * 16 + n] = acc[s][r];
}

// ---------------- Kernel 2: exclusive prefix sum over chunks (per batch) ----------------
__global__ void __launch_bounds__(256)
prefix_kernel(const float* __restrict__ G, float* __restrict__ Gpre) {
  const int b = blockIdx.x;
  const float* Gb = G + (size_t)b * NC * (NDIM * NDIM);
  float* Pb = Gpre + (size_t)b * NC * (NDIM * NDIM);
  float run[16];
#pragma unroll
  for (int e = 0; e < 16; ++e) run[e] = 0.f;
  for (int c = 0; c < NC; ++c) {
#pragma unroll
    for (int e = 0; e < 16; ++e) {
      const int idx = e * 256 + threadIdx.x;
      Pb[c * (NDIM * NDIM) + idx] = run[e];
      run[e] += Gb[c * (NDIM * NDIM) + idx];
    }
  }
}

// ---------------- Kernel 3: main chunked causal kernel ----------------
__global__ void __launch_bounds__(256)
chunk_kernel(const float* __restrict__ rp, const float* __restrict__ Q,
             const float* __restrict__ E, const float* __restrict__ Gpre,
             float* __restrict__ out) {
  alignas(16) __shared__ __bf16 Rl [64 * LDK];  // R_c row-major  [t][i]
  alignas(16) __shared__ __bf16 Qm [64 * LDK];  // Q_h row-major  [i][j]
  alignas(16) __shared__ __bf16 Qt [64 * LDK];  // Q_h^T          [j][i]
  alignas(16) __shared__ __bf16 Em [64 * LDK];  // E_h row-major  [i][j]
  alignas(16) __shared__ __bf16 Gpl[64 * LDK];  // Gpre (symmetric)
  alignas(16) __shared__ __bf16 Yl [64 * LDK];  // Y = Q*Gpre
  alignas(16) __shared__ __bf16 Pl [64 * LDK];  // P = R*Q
  alignas(16) __shared__ __bf16 Sl [64 * LDK];  // S = tril(P R^T)
  alignas(16) __shared__ __bf16 Vtl[64 * LDK];  // V^T, V = R*E^T
  alignas(16) __shared__ __bf16 Wtl[64 * LDK];  // W^T, W = sum_h Q Gpre E^T

  const int bc = blockIdx.x, b = bc / NC, c = bc % NC;
  const int tid = threadIdx.x;
  const int wave = tid >> 5, lane = tid & 31;
  const int mt = wave & 3, ntb = (wave >> 2) * 2;
  const int n = lane & 15, hi = lane >> 4;
  const v8f vz = {0.f, 0.f, 0.f, 0.f, 0.f, 0.f, 0.f, 0.f};

  const float* Rg = rp + ((size_t)b * NT_ + (size_t)c * CHUNK) * NDIM;
  const float* Gg = Gpre + ((size_t)b * NC + c) * (NDIM * NDIM);

  for (int idx = tid; idx < NDIM * NDIM; idx += 256) {
    const int rI = idx >> 6, cI = idx & 63;
    Rl [rI * LDK + cI] = (__bf16)Rg[idx];
    Gpl[rI * LDK + cI] = (__bf16)Gg[idx];
  }

  v8f accO[2] = {vz, vz};
  v8f accW[2] = {vz, vz};

  for (int h = 0; h < N_H; ++h) {
    __syncthreads();
    const float* Qg = Q + (size_t)h * NDIM * NDIM;
    const float* Eg = E + (size_t)h * NDIM * NDIM;
    for (int idx = tid; idx < NDIM * NDIM; idx += 256) {
      const int i = idx >> 6, j = idx & 63;
      const float q = Qg[idx];
      Qm[i * LDK + j] = (__bf16)q;
      Qt[j * LDK + i] = (__bf16)q;
      Em[i * LDK + j] = (__bf16)Eg[idx];
    }
    __syncthreads();

    // Phase b: P = R*Q (B from Qt) ; V = R*E^T (B from Em) ; Y = Q*Gpre (B image = Gpl, symmetric)
    // k-outer so one A fragment feeds multiple WMMAs.
    {
      v8f accP[2] = {vz, vz}, accV[2] = {vz, vz}, accY[2] = {vz, vz};
#pragma unroll
      for (int k0 = 0; k0 < 64; k0 += 32) {
        v16bf aR = frag_a(Rl, mt * 16, k0, lane);
        v16bf aQ = frag_a(Qm, mt * 16, k0, lane);
#pragma unroll
        for (int s = 0; s < 2; ++s) {
          const int nt = ntb + s;
          accP[s] = wmma_bf16(aR, frag_bt(Qt,  nt * 16, k0, lane), accP[s]);
          accV[s] = wmma_bf16(aR, frag_bt(Em,  nt * 16, k0, lane), accV[s]);
          accY[s] = wmma_bf16(aQ, frag_bt(Gpl, nt * 16, k0, lane), accY[s]);
        }
      }
#pragma unroll
      for (int s = 0; s < 2; ++s) {
        store_bf16  (Pl,  mt, ntb + s, lane, accP[s]);
        store_bf16  (Yl,  mt, ntb + s, lane, accY[s]);
        store_bf16_t(Vtl, mt, ntb + s, lane, accV[s]);
      }
    }
    __syncthreads();

    // Phase c: W += Y*E^T (register-resident) ; S = tril(P * R^T) (Bt image for R^T is Rl)
    {
      v8f accS[2] = {vz, vz};
#pragma unroll
      for (int k0 = 0; k0 < 64; k0 += 32) {
        v16bf aY = frag_a(Yl, mt * 16, k0, lane);
        v16bf aP = frag_a(Pl, mt * 16, k0, lane);
#pragma unroll
        for (int s = 0; s < 2; ++s) {
          const int nt = ntb + s;
          accW[s] = wmma_bf16(aY, frag_bt(Em, nt * 16, k0, lane), accW[s]);
          accS[s] = wmma_bf16(aP, frag_bt(Rl, nt * 16, k0, lane), accS[s]);
        }
      }
#pragma unroll
      for (int s = 0; s < 2; ++s) {
        const int nt = ntb + s;
#pragma unroll
        for (int r = 0; r < 8; ++r) {
          const int t = mt * 16 + r + 8 * hi;
          const int u = nt * 16 + n;
          if (t < u) accS[s][r] = 0.0f;    // causal: keep t >= u
        }
        store_bf16(Sl, mt, nt, lane, accS[s]);
      }
    }
    __syncthreads();

    // Phase d: accO += S * V   (Bt image = Vtl)
#pragma unroll
    for (int k0 = 0; k0 < 64; k0 += 32) {
      v16bf aS = frag_a(Sl, mt * 16, k0, lane);
#pragma unroll
      for (int s = 0; s < 2; ++s)
        accO[s] = wmma_bf16(aS, frag_bt(Vtl, (ntb + s) * 16, k0, lane), accO[s]);
    }
  }

  __syncthreads();
#pragma unroll
  for (int s = 0; s < 2; ++s) store_bf16_t(Wtl, mt, ntb + s, lane, accW[s]);
  __syncthreads();

  // accO += R * W   (Bt image = Wtl) ; then store fp32 output
#pragma unroll
  for (int k0 = 0; k0 < 64; k0 += 32) {
    v16bf aR = frag_a(Rl, mt * 16, k0, lane);
#pragma unroll
    for (int s = 0; s < 2; ++s)
      accO[s] = wmma_bf16(aR, frag_bt(Wtl, (ntb + s) * 16, k0, lane), accO[s]);
  }

  float* Og = out + ((size_t)b * NT_ + (size_t)c * CHUNK) * NDIM;
#pragma unroll
  for (int s = 0; s < 2; ++s)
#pragma unroll
    for (int r = 0; r < 8; ++r) {
      const int t = mt * 16 + r + 8 * hi;
      Og[t * NDIM + (ntb + s) * 16 + n] = accO[s][r];
    }
}

// ---------------- Host launcher ----------------
extern "C" void kernel_launch(void* const* d_in, const int* in_sizes, int n_in,
                              void* d_out, int out_size, void* d_ws, size_t ws_size,
                              hipStream_t stream) {
  const float* rp = (const float*)d_in[0];   // (8, 2048, 64)
  const float* Q  = (const float*)d_in[1];   // (8, 64, 64)
  const float* E  = (const float*)d_in[2];   // (8, 64, 64)
  float* out = (float*)d_out;                // (8, 2048, 64) f32

  // Workspace: G (per-chunk Gram) then Gpre (exclusive prefix), each 8*32*64*64 f32 = 4 MB
  float* G    = (float*)d_ws;
  float* Gpre = G + (size_t)D_B * NC * NDIM * NDIM;

  g_kernel     <<<D_B * NC, 256, 0, stream>>>(rp, G);
  prefix_kernel<<<D_B,      256, 0, stream>>>(G, Gpre);
  chunk_kernel <<<D_B * NC, 256, 0, stream>>>(rp, Q, E, Gpre, out);
}
